// Model_4329327035158
// MI455X (gfx1250) — matrix operand order repaired and checked by reading, not compiled, not run on hardware
//
#include <hip/hip_runtime.h>
#include <hip/hip_bf16.h>
#include <math.h>

#define D 256
#define WIN 3
#define WLEN 7
#define ROWS_PER_WG 64
#define BLOCK 128
#define HALO 3
#define STAGE (ROWS_PER_WG + 2 * HALO)   // 70 staged rows
#define XSTRIDE 264                      // ushorts per bf16 row; 528B (16B-aligned stride)
#define XFSTRIDE 260                     // floats per fp32 row; 1040B (16B-aligned stride)

typedef __attribute__((ext_vector_type(2)))  float        v2f;
typedef __attribute__((ext_vector_type(8)))  float        v8f;
typedef __attribute__((ext_vector_type(8)))  unsigned int v8u;
typedef __attribute__((ext_vector_type(16))) __bf16       v16bf;
typedef __attribute__((ext_vector_type(4)))  int          v4i;

typedef __attribute__((address_space(1))) v4i GV4;       // global int4
typedef __attribute__((address_space(3))) v4i LV4;       // LDS int4

__device__ __forceinline__ unsigned short f2bf(float f) {
    unsigned u = __builtin_bit_cast(unsigned, f);
    u += 0x7FFFu + ((u >> 16) & 1u);     // round-to-nearest-even
    return (unsigned short)(u >> 16);
}

__global__ __launch_bounds__(BLOCK)
void slwin_attn_kernel(const float* __restrict__ x,
                       const float* __restrict__ Wp,
                       const float* __restrict__ bp,
                       float* __restrict__ out, int N) {
    __shared__ __align__(16) float          xf [STAGE * XFSTRIDE]; // fp32 staged rows
    __shared__ __align__(16) unsigned short xbf[STAGE * XSTRIDE];  // bf16 staged rows
    __shared__ __align__(16) float wsS[2 * D];
    __shared__ float normS[STAGE];
    __shared__ float ywS[STAGE];
    __shared__ float ycS[STAGE];
    __shared__ float Gs[4][16][34];          // per-wave 16x32 Gram band (+pad)
    __shared__ float attnS[ROWS_PER_WG * 8];

    const int tid  = threadIdx.x;
    const int g0   = blockIdx.x * ROWS_PER_WG;
    const int lane = tid & 31;
    const int wave = tid >> 5;
    const int half = lane >> 4;
    const int lrow = lane & 15;

    // ---- async-stage x rows [g0-3, g0+67) fp32 -> LDS; zero-fill outside [0,N)
    for (int idx = tid; idx < STAGE * (D / 4); idx += BLOCK) {
        int m = idx / (D / 4), c4 = idx % (D / 4);
        int grow = g0 - HALO + m;
        float* lp = &xf[m * XFSTRIDE + c4 * 4];
        if (grow >= 0 && grow < N) {
#if __has_builtin(__builtin_amdgcn_global_load_async_to_lds_b128)
            __builtin_amdgcn_global_load_async_to_lds_b128(
                (GV4*)(x + (long)grow * D + c4 * 4), (LV4*)lp, 0, 0);
#else
            *(float4*)lp = *(const float4*)(x + (long)grow * D + c4 * 4);
#endif
        } else {
            *(float4*)lp = make_float4(0.f, 0.f, 0.f, 0.f);
        }
    }
    // ---- async-stage W (2*D floats == BLOCK float4s)
    {
        int c4 = tid * 4;
#if __has_builtin(__builtin_amdgcn_global_load_async_to_lds_b128)
        __builtin_amdgcn_global_load_async_to_lds_b128(
            (GV4*)(Wp + c4), (LV4*)&wsS[c4], 0, 0);
#else
        *(float4*)&wsS[c4] = *(const float4*)(Wp + c4);
#endif
    }
    const float wsig = Wp[2 * D];
    const float bias = bp[0];
#if __has_builtin(__builtin_amdgcn_s_wait_asynccnt)
    __builtin_amdgcn_s_wait_asynccnt(0);
#else
    asm volatile("s_wait_asynccnt 0" ::: "memory");
#endif
    __syncthreads();

    // ---- bf16 shadow copy for the Gram WMMA (LDS -> LDS convert)
    for (int idx = tid; idx < STAGE * (D / 4); idx += BLOCK) {
        int m = idx / (D / 4), c4 = idx % (D / 4);
        float4 v = *(const float4*)&xf[m * XFSTRIDE + c4 * 4];
        int base = m * XSTRIDE + c4 * 4;
        xbf[base + 0] = f2bf(v.x); xbf[base + 1] = f2bf(v.y);
        xbf[base + 2] = f2bf(v.z); xbf[base + 3] = f2bf(v.w);
    }
    __syncthreads();

    // ---- exact fp32 norms + x.w_win + x.w_cen (16 threads/row, shfl reduce; LDS reads)
    {
        const int slot = tid >> 4, tt = tid & 15;
        for (int it = 0; it < (STAGE + 7) / 8; ++it) {
            int m = it * 8 + slot;
            float ss = 0.f, dw = 0.f, dc = 0.f;
            if (m < STAGE) {
                const float* xr = &xf[m * XFSTRIDE];
                #pragma unroll
                for (int j = 0; j < 4; ++j) {
                    int col = tt * 4 + j * 64;
                    float4 v = *(const float4*)(xr + col);
                    ss += v.x * v.x + v.y * v.y + v.z * v.z + v.w * v.w;
                    dw += v.x * wsS[col] + v.y * wsS[col + 1] + v.z * wsS[col + 2] + v.w * wsS[col + 3];
                    dc += v.x * wsS[D + col] + v.y * wsS[D + col + 1] + v.z * wsS[D + col + 2] + v.w * wsS[D + col + 3];
                }
            }
            #pragma unroll
            for (int msk = 8; msk >= 1; msk >>= 1) {   // stays inside 16-lane group
                ss += __shfl_xor(ss, msk, 32);
                dw += __shfl_xor(dw, msk, 32);
                dc += __shfl_xor(dc, msk, 32);
            }
            if (tt == 0 && m < STAGE) { normS[m] = sqrtf(ss); ywS[m] = dw; ycS[m] = dc; }
        }
    }
    __syncthreads();

    // ---- banded Gram via bf16 WMMA: wave handles rows R..R+15, cols R-3..R+28 (2 tiles)
    {
        int ma  = (HALO + 16 * wave + lrow) * XSTRIDE;                 // A row for this lane
        int mb0 = (16 * wave + lrow) * XSTRIDE;                        // B tile0 col-row
        int mb1 = (min(16 * wave + 16 + lrow, STAGE - 1)) * XSTRIDE;   // B tile1 (clamped; excess unused)
        v8f c1 = {}; v8f c2 = {};
        #pragma unroll
        for (int kb = 0; kb < D; kb += 32) {
            // A frag: 16 contiguous bf16 at K=kb+16*half (matches B)
            uint4 qa1 = *(const uint4*)&xbf[ma + kb + 16 * half];
            uint4 qa2 = *(const uint4*)&xbf[ma + kb + 16 * half + 8];
            v8u ua = { qa1.x, qa1.y, qa1.z, qa1.w, qa2.x, qa2.y, qa2.z, qa2.w };
            v16bf a = __builtin_bit_cast(v16bf, ua);
            // B frag: 16 contiguous bf16 at K=kb+16*half                (contiguous k order, same as A)
            const uint4* pb0 = (const uint4*)&xbf[mb0 + kb + 16 * half];
            uint4 q0 = pb0[0], q1 = pb0[1];
            v8u ub0 = { q0.x, q0.y, q0.z, q0.w, q1.x, q1.y, q1.z, q1.w };
            v16bf b0 = __builtin_bit_cast(v16bf, ub0);
            const uint4* pb1 = (const uint4*)&xbf[mb1 + kb + 16 * half];
            uint4 r0 = pb1[0], r1 = pb1[1];
            v8u ub1 = { r0.x, r0.y, r0.z, r0.w, r1.x, r1.y, r1.z, r1.w };
            v16bf b1 = __builtin_bit_cast(v16bf, ub1);

            c1 = __builtin_amdgcn_wmma_f32_16x16x32_bf16(false, a, false, b0, (short)0, c1, false, false);
            c2 = __builtin_amdgcn_wmma_f32_16x16x32_bf16(false, a, false, b1, (short)0, c2, false, false);
        }
        #pragma unroll
        for (int v = 0; v < 8; ++v) {                 // C/D layout: M=v+8*half, N=lrow
            int M = v + 8 * half;
            Gs[wave][M][lrow]      = c1[v];
            Gs[wave][M][16 + lrow] = c2[v];
        }
    }
    __syncthreads();

    // ---- scores + masked softmax (one thread per row)
    if (tid < ROWS_PER_WG && g0 + tid < N) {
        int i = tid, wv = i >> 4, ii = i & 15;
        float na = fmaxf(normS[i + HALO], 1e-12f);
        float yc = ycS[i + HALO];
        float sc[WLEN]; float mx = -INFINITY;
        #pragma unroll
        for (int w = 0; w < WLEN; ++w) {
            int ng = g0 + i + w - WIN;
            float nb   = fmaxf(normS[i + w], 1e-12f);
            float cosv = Gs[wv][ii][ii + w] / (na * nb);
            float sig  = 1.f / (1.f + __expf(-cosv));
            float s    = ywS[i + w] + yc + sig * wsig + bias;
            s = (ng >= 0 && ng < N) ? s : -INFINITY;
            sc[w] = s; mx = fmaxf(mx, s);
        }
        float sum = 0.f, e[WLEN];
        #pragma unroll
        for (int w = 0; w < WLEN; ++w) { e[w] = __expf(sc[w] - mx); sum += e[w]; }
        float inv = 1.f / sum;
        #pragma unroll
        for (int w = 0; w < WLEN; ++w) attnS[i * 8 + w] = e[w] * inv;
    }
    __syncthreads();

    // ---- out[R:R+16,:] = attn-band(16x24) @ X(24x256) via fp32 WMMA 16x16x4 (B from LDS)
    {
        int n0w = g0 + 16 * wave;
        v2f afrag[6];
        int mrow[12];
        #pragma unroll
        for (int s = 0; s < 6; ++s) {                  // f32 A layout: K pair at 2*half
            int k0 = 4 * s + 2 * half;
            int w0 = k0 - lrow, w1 = k0 + 1 - lrow;
            v2f af;
            af.x = (w0 >= 0 && w0 < WLEN) ? attnS[(16 * wave + lrow) * 8 + w0] : 0.f;
            af.y = (w1 >= 0 && w1 < WLEN) ? attnS[(16 * wave + lrow) * 8 + w1] : 0.f;
            afrag[s] = af;
            #pragma unroll
            for (int j = 0; j < 2; ++j) {
                int kr = k0 + j;                       // staged-local row 16*wave + kr
                mrow[2 * s + j] = min(16 * wave + kr, STAGE - 1) * XFSTRIDE;
            }
        }
        #pragma unroll 4
        for (int db = 0; db < D; db += 16) {
            v8f acc = {};
            #pragma unroll
            for (int s = 0; s < 6; ++s) {              // f32 B layout: K pair at 2*half, N=lrow
                v2f bf;
                bf.x = xf[mrow[2 * s]     + db + lrow];
                bf.y = xf[mrow[2 * s + 1] + db + lrow];
                acc = __builtin_amdgcn_wmma_f32_16x16x4_f32(false, afrag[s], false, bf,
                                                            (short)0, acc, false, false);
            }
            #pragma unroll
            for (int v = 0; v < 8; ++v) {
                int M = v + 8 * half;
                if (n0w + M < N) out[(long)(n0w + M) * D + db + lrow] = acc[v];
            }
        }
    }
}

extern "C" void kernel_launch(void* const* d_in, const int* in_sizes, int n_in,
                              void* d_out, int out_size, void* d_ws, size_t ws_size,
                              hipStream_t stream) {
    const float* x = (const float*)d_in[0];
    const float* W = (const float*)d_in[1];
    const float* b = (const float*)d_in[2];
    float* out = (float*)d_out;
    int N = in_sizes[0] / D;
    int nblocks = (N + ROWS_PER_WG - 1) / ROWS_PER_WG;
    hipLaunchKernelGGL(slwin_attn_kernel, dim3(nblocks), dim3(BLOCK), 0, stream,
                       x, W, b, out, N);
}
